// SelfAttention_16750372454812
// MI455X (gfx1250) — matrix operand order repaired and checked by reading, not compiled, not run on hardware
//
#include <hip/hip_runtime.h>
#include <hip/hip_bf16.h>

// ---------------- problem constants (match reference) ----------------
#define BB   2
#define SS   2048
#define DD   4096
#define NH   32
#define NKV  8
#define HD   128
#define NREP (NH / NKV)
#define HD2  (HD / 2)
#define MM   (BB * SS)        // 4096 rows
#define DQ   (NH * HD)        // 4096
#define DKV  (NKV * HD)       // 1024
#define SCALE 0.08838834764831845f   // 1/sqrt(128)

typedef __bf16 bf16;
typedef __attribute__((ext_vector_type(16))) __bf16 v16bf;
typedef __attribute__((ext_vector_type(8)))  __bf16 v8bf;
typedef __attribute__((ext_vector_type(8)))  float  v8f;

union V16 { v16bf v; v8bf h[2]; };

__device__ __forceinline__ v8f wmma_bf16(v16bf a, v16bf b, v8f c) {
  // (neg_a, A, neg_b, B, c_mod, C, reuse_a, reuse_b)
  return __builtin_amdgcn_wmma_f32_16x16x32_bf16(false, a, false, b,
                                                 (short)0, c, false, false);
}

// LDS byte offset of a __shared__ object: flat shared addresses carry the LDS
// offset in addr[31:0] (aperture truncation rule, ISA 10.2).
__device__ __forceinline__ unsigned lds_of(const void* p) {
  return (unsigned)(size_t)p;
}

// gfx1250 async global->LDS DMA (ASYNCcnt-tracked, bypasses VGPRs)
__device__ __forceinline__ void async_ld_b128(unsigned lds_off, const void* gptr) {
  asm volatile("global_load_async_to_lds_b128 %0, %1, off"
               :: "v"(lds_off), "v"((unsigned long long)(size_t)gptr)
               : "memory");
}

__device__ __forceinline__ void wait_async_le4() {
#if __has_builtin(__builtin_amdgcn_s_wait_asynccnt)
  __builtin_amdgcn_s_wait_asynccnt(4);
#else
  asm volatile("s_wait_asynccnt 4" ::: "memory");
#endif
}
__device__ __forceinline__ void wait_async_0() {
#if __has_builtin(__builtin_amdgcn_s_wait_asynccnt)
  __builtin_amdgcn_s_wait_asynccnt(0);
#else
  asm volatile("s_wait_asynccnt 0" ::: "memory");
#endif
}

// ---------------- f32 -> bf16 conversion ----------------
__global__ __launch_bounds__(256) void cvt_f32_bf16(const float* __restrict__ src,
                                                    bf16* __restrict__ dst,
                                                    long long n) {
  long long i = (long long)blockIdx.x * blockDim.x + threadIdx.x;
  long long stride = (long long)gridDim.x * blockDim.x;
  for (; i < n; i += stride) dst[i] = (bf16)src[i];
}

// ---------------- tiled WMMA GEMM:  C[M,N] (f32) = A[M,K] (bf16) * W[N,K]^T (bf16)
// 128x128 macro-tile, 8 waves (2 M-groups x 4 N-groups), each wave owns 64x32.
// K-step 32, double-buffered LDS filled by async global->LDS DMA.
__global__ __launch_bounds__(256) void gemm_bf16_nt(const bf16* __restrict__ A,
                                                    const bf16* __restrict__ W,
                                                    float* __restrict__ C,
                                                    int M, int N, int K) {
  __shared__ bf16 As[2][128][32];
  __shared__ bf16 Ws[2][128][32];

  const int tid  = threadIdx.x;
  const int lane = tid & 31;
  const int wave = tid >> 5;       // 0..7
  const int wy   = wave >> 2;      // 0..1 : 64-row group
  const int wx   = wave & 3;       // 0..3 : 32-col group (two 16x16 tiles wide)
  const int half = lane >> 4;      // 0/1
  const int l16  = lane & 15;

  const int m0 = blockIdx.y * 128;
  const int n0 = blockIdx.x * 128;

  // cooperative async fill: 128 rows x 32 cols per array; 4 b128 per thread
  const int r0 = tid >> 2;         // 0..63
  const int sg = (tid & 3) * 8;    // 0,8,16,24

  v8f acc[4][2];
#pragma unroll
  for (int i = 0; i < 4; ++i)
#pragma unroll
    for (int j = 0; j < 2; ++j) acc[i][j] = (v8f){};

  auto issue = [&](int buf, int kt) {
    async_ld_b128(lds_of(&As[buf][r0][sg]),      &A[(size_t)(m0 + r0)      * K + kt + sg]);
    async_ld_b128(lds_of(&As[buf][r0 + 64][sg]), &A[(size_t)(m0 + r0 + 64) * K + kt + sg]);
    async_ld_b128(lds_of(&Ws[buf][r0][sg]),      &W[(size_t)(n0 + r0)      * K + kt + sg]);
    async_ld_b128(lds_of(&Ws[buf][r0 + 64][sg]), &W[(size_t)(n0 + r0 + 64) * K + kt + sg]);
  };

  issue(0, 0);
  int buf = 0;
  for (int kt = 0; kt < K; kt += 32) {
    const bool hasnext = (kt + 32) < K;
    if (hasnext) {
      issue(buf ^ 1, kt + 32);   // prefetch next K-panel into the other buffer
      wait_async_le4();          // async loads complete in order: cur buf's 4 done
    } else {
      wait_async_0();
    }
    __syncthreads();             // all waves' DMA for cur buf has landed

    // B fragments (32x16): lane col = l16, K chunks [half*8,+8) and [16+half*8,+8)
    V16 bfr[2];
#pragma unroll
    for (int j = 0; j < 2; ++j) {
      const bf16* wr = &Ws[buf][wx * 32 + j * 16 + l16][0];
      bfr[j].h[0] = *(const v8bf*)(wr + half * 8);
      bfr[j].h[1] = *(const v8bf*)(wr + 16 + half * 8);
    }
    // A fragments (16x32): lane row = l16, K chunks [half*8,+8) and [16+half*8,+8)
#pragma unroll
    for (int i = 0; i < 4; ++i) {
      V16 af;
      const bf16* ar = &As[buf][wy * 64 + i * 16 + l16][0];
      af.h[0] = *(const v8bf*)&ar[half * 8];
      af.h[1] = *(const v8bf*)&ar[16 + half * 8];
      acc[i][0] = wmma_bf16(af.v, bfr[0].v, acc[i][0]);
      acc[i][1] = wmma_bf16(af.v, bfr[1].v, acc[i][1]);
    }
    __syncthreads();             // done reading cur buf before it is refilled
    buf ^= 1;
  }

  // D layout: vgpr r -> row = r + 8*half, col = l16
#pragma unroll
  for (int i = 0; i < 4; ++i) {
    const int rbase = m0 + wy * 64 + i * 16 + half * 8;
#pragma unroll
    for (int r = 0; r < 8; ++r) {
      C[(size_t)(rbase + r) * N + n0 + wx * 32 + l16]      = acc[i][0][r];
      C[(size_t)(rbase + r) * N + n0 + wx * 32 + 16 + l16] = acc[i][1][r];
    }
  }
}

// ---------------- RoPE on K and V (reference ropes both, not Q) + bf16 repack
// Kf,Vf: [B*S, NKV*HD] f32   cos,sin: [S, HD/2] f32
// Kb: [B][NKV][S][HD] bf16   Vt: [B][NKV][HD][S] bf16 (transposed for P*V B-operand)
__global__ __launch_bounds__(256) void rope_pack(const float* __restrict__ Kf,
                                                 const float* __restrict__ Vf,
                                                 const float* __restrict__ cosT,
                                                 const float* __restrict__ sinT,
                                                 bf16* __restrict__ Kb,
                                                 bf16* __restrict__ Vt) {
  int idx = blockIdx.x * blockDim.x + threadIdx.x;   // B*S*NKV*HD2 threads
  const int i   = idx % HD2;
  const int kvh = (idx / HD2) % NKV;
  const int s   = (idx / (HD2 * NKV)) % SS;
  const int b   = idx / (HD2 * NKV * SS);

  const float c  = cosT[s * HD2 + i];
  const float sn = sinT[s * HD2 + i];

  const size_t src = ((size_t)(b * SS + s) * NKV + kvh) * HD + 2 * i;
  const float kr = Kf[src], ki = Kf[src + 1];
  const float vr = Vf[src], vi = Vf[src + 1];

  const float kor = kr * c - ki * sn, koi = kr * sn + ki * c;
  const float vor = vr * c - vi * sn, voi = vr * sn + vi * c;

  const size_t kdst = ((size_t)(b * NKV + kvh) * SS + s) * HD + 2 * i;
  Kb[kdst]     = (bf16)kor;
  Kb[kdst + 1] = (bf16)koi;

  const size_t vbase = (size_t)(b * NKV + kvh) * HD * SS;
  Vt[vbase + (size_t)(2 * i)     * SS + s] = (bf16)vor;
  Vt[vbase + (size_t)(2 * i + 1) * SS + s] = (bf16)voi;
}

// ---------------- flash attention: 1 wave = 16 q rows, block = 64 q rows of one (b,h)
// Qb: [B,S,NH,HD]  Kb: [B,NKV,S,HD]  Vt: [B,NKV,HD,S]  Ob: [B,S,NH,HD] (bf16)
__global__ __launch_bounds__(128) void attn_kernel(const bf16* __restrict__ Qb,
                                                   const bf16* __restrict__ Kb,
                                                   const bf16* __restrict__ Vt,
                                                   bf16* __restrict__ Ob) {
  __shared__ bf16 Pl[4][16][32];   // per-wave P tile for D-layout -> A-layout transpose

  const int lane = threadIdx.x & 31;
  const int wave = threadIdx.x >> 5;      // 0..3
  const int half = lane >> 4;
  const int l16  = lane & 15;

  const int q0  = blockIdx.x * 64 + wave * 16;
  const int h   = blockIdx.y;
  const int b   = blockIdx.z;
  const int kvh = h / NREP;

  // Q fragments: 4 chunks of K=32 covering HD=128 (Q is NOT roped in reference)
  const bf16* qrow = Qb + ((size_t)(b * SS + q0 + l16) * NH + h) * HD;
  V16 qf[4];
#pragma unroll
  for (int c = 0; c < 4; ++c) {
    qf[c].h[0] = *(const v8bf*)&qrow[c * 32 + half * 8];
    qf[c].h[1] = *(const v8bf*)&qrow[c * 32 + 16 + half * 8];
  }

  v8f acc[8];
#pragma unroll
  for (int t = 0; t < 8; ++t) acc[t] = (v8f){};
  float mrow[8], lrow[8];
#pragma unroll
  for (int r = 0; r < 8; ++r) { mrow[r] = -1e30f; lrow[r] = 0.f; }

  const bf16* Kbase = Kb + (size_t)(b * NKV + kvh) * SS * HD;
  const bf16* Vbase = Vt + (size_t)(b * NKV + kvh) * HD * SS;

  for (int kt = 0; kt < SS; kt += 32) {
    // ---- S = Q * K^T : two 16x16 tiles (keys kt..+15, kt+16..+31)
    v8f st0 = (v8f){}, st1 = (v8f){};
#pragma unroll
    for (int c = 0; c < 4; ++c) {
      V16 kf0, kf1;
      const bf16* kr0 = Kbase + (size_t)(kt + l16)      * HD + c * 32;
      const bf16* kr1 = Kbase + (size_t)(kt + 16 + l16) * HD + c * 32;
      kf0.h[0] = *(const v8bf*)(kr0 + half * 8);       kf0.h[1] = *(const v8bf*)(kr0 + 16 + half * 8);
      kf1.h[0] = *(const v8bf*)(kr1 + half * 8);       kf1.h[1] = *(const v8bf*)(kr1 + 16 + half * 8);
      st0 = wmma_bf16(qf[c].v, kf0.v, st0);
      st1 = wmma_bf16(qf[c].v, kf1.v, st1);
    }

    // ---- online softmax: S-tile and acc share (row=r+8*half, col=l16) layout
#pragma unroll
    for (int r = 0; r < 8; ++r) {
      float s0 = st0[r] * SCALE;
      float s1 = st1[r] * SCALE;
      float mx = fmaxf(s0, s1);
#pragma unroll
      for (int off = 8; off >= 1; off >>= 1)
        mx = fmaxf(mx, __shfl_xor(mx, off, 16));
      const float mnew  = fmaxf(mrow[r], mx);
      const float alpha = __expf(mrow[r] - mnew);
      const float p0    = __expf(s0 - mnew);
      const float p1    = __expf(s1 - mnew);
      float psum = p0 + p1;
#pragma unroll
      for (int off = 8; off >= 1; off >>= 1)
        psum += __shfl_xor(psum, off, 16);
      lrow[r] = lrow[r] * alpha + psum;
      mrow[r] = mnew;
#pragma unroll
      for (int t = 0; t < 8; ++t) acc[t][r] *= alpha;
      Pl[wave][r + 8 * half][l16]      = (bf16)p0;
      Pl[wave][r + 8 * half][16 + l16] = (bf16)p1;
    }
    __syncthreads();

    // ---- P (A-operand, 16x32) from LDS; V (B-operand) contiguous key-runs
    V16 pf;
    const bf16* prow = &Pl[wave][l16][0];
    pf.h[0] = *(const v8bf*)&prow[half * 8];
    pf.h[1] = *(const v8bf*)&prow[16 + half * 8];
#pragma unroll
    for (int t = 0; t < 8; ++t) {
      V16 vf;
      const bf16* vcol = Vbase + (size_t)(t * 16 + l16) * SS + kt;
      vf.h[0] = *(const v8bf*)(vcol + half * 8);
      vf.h[1] = *(const v8bf*)(vcol + 16 + half * 8);
      acc[t] = wmma_bf16(pf.v, vf.v, acc[t]);
    }
    __syncthreads();
  }

  // ---- normalize and store (bf16, [B,S,NH,HD])
#pragma unroll
  for (int r = 0; r < 8; ++r) {
    const float inv = 1.0f / lrow[r];
    const size_t row = (size_t)q0 + r + 8 * half;
    bf16* orow = Ob + ((size_t)(b * SS) + row) * NH * HD + (size_t)h * HD;
#pragma unroll
    for (int t = 0; t < 8; ++t)
      orow[t * 16 + l16] = (bf16)(acc[t][r] * inv);
  }
}

// ---------------- host-side orchestration ----------------
extern "C" void kernel_launch(void* const* d_in, const int* in_sizes, int n_in,
                              void* d_out, int out_size, void* d_ws, size_t ws_size,
                              hipStream_t stream) {
  (void)in_sizes; (void)n_in; (void)out_size; (void)ws_size;
  const float* x  = (const float*)d_in[0];
  // d_in[1] = start_pos (unused by reference rope table)
  const float* fc = (const float*)d_in[2];
  const float* fs = (const float*)d_in[3];
  const float* wq = (const float*)d_in[4];
  const float* wk = (const float*)d_in[5];
  const float* wv = (const float*)d_in[6];
  const float* wo = (const float*)d_in[7];
  float* out = (float*)d_out;

  char* p = (char*)d_ws;
  auto alloc = [&](size_t bytes) -> char* {
    char* r = p; p += (bytes + 255) & ~(size_t)255; return r;
  };

  const size_t nX  = (size_t)MM * DD;    // 16.7M
  const size_t nQ  = (size_t)MM * DQ;    // 16.7M
  const size_t nKV = (size_t)MM * DKV;   // 4.19M

  bf16* xb    = (bf16*)alloc(nX  * 2);
  bf16* wqb   = (bf16*)alloc((size_t)DQ  * DD * 2);
  bf16* wkb   = (bf16*)alloc((size_t)DKV * DD * 2);
  bf16* wvb   = (bf16*)alloc((size_t)DKV * DD * 2);
  bf16* wob   = (bf16*)alloc((size_t)DD  * DQ * 2);
  float* Qf   = (float*)alloc(nQ  * 4);
  float* Kf   = (float*)alloc(nKV * 4);
  float* Vf   = (float*)alloc(nKV * 4);
  bf16* Qbf   = (bf16*)alloc(nQ  * 2);
  bf16* Kbf   = (bf16*)alloc(nKV * 2);
  bf16* Vtbf  = (bf16*)alloc(nKV * 2);
  bf16* attnb = (bf16*)alloc(nQ  * 2);

  // 1) convert inputs to bf16
  cvt_f32_bf16<<<2048, 256, 0, stream>>>(x,  xb,  (long long)nX);
  cvt_f32_bf16<<<2048, 256, 0, stream>>>(wq, wqb, (long long)DQ  * DD);
  cvt_f32_bf16<<<1024, 256, 0, stream>>>(wk, wkb, (long long)DKV * DD);
  cvt_f32_bf16<<<1024, 256, 0, stream>>>(wv, wvb, (long long)DKV * DD);
  cvt_f32_bf16<<<2048, 256, 0, stream>>>(wo, wob, (long long)DD  * DQ);

  // 2) Q/K/V projections (WMMA GEMM, C = A * W^T), 128x128 macro-tiles
  gemm_bf16_nt<<<dim3(DQ  / 128, MM / 128), 256, 0, stream>>>(xb, wqb, Qf, MM, DQ,  DD);
  gemm_bf16_nt<<<dim3(DKV / 128, MM / 128), 256, 0, stream>>>(xb, wkb, Kf, MM, DKV, DD);
  gemm_bf16_nt<<<dim3(DKV / 128, MM / 128), 256, 0, stream>>>(xb, wvb, Vf, MM, DKV, DD);

  // 3) Q -> bf16 ; RoPE(K,V) + repack (K row-major per head, V transposed)
  cvt_f32_bf16<<<2048, 256, 0, stream>>>(Qf, Qbf, (long long)nQ);
  rope_pack<<<(BB * SS * NKV * HD2) / 256, 256, 0, stream>>>(Kf, Vf, fc, fs, Kbf, Vtbf);

  // 4) flash attention
  attn_kernel<<<dim3(SS / 64, NH, BB), 128, 0, stream>>>(Qbf, Kbf, Vtbf, attnb);

  // 5) output projection straight into d_out (f32)
  gemm_bf16_nt<<<dim3(DD / 128, MM / 128), 256, 0, stream>>>(attnb, wob, out, MM, DD, DQ);
}
